// ExecutableModel_45878840656240
// MI455X (gfx1250) — matrix operand order repaired and checked by reading, not compiled, not run on hardware
//
#include <hip/hip_runtime.h>
#include <hip/hip_bf16.h>

// ---------------- model dims ----------------
constexpr int B = 2, S = 1024, D = 1024, H = 16, L = 4, F = 4096, V = 32000;
constexpr int DH = D / H;          // 64
constexpr int M = B * S;           // 2048 rows
constexpr int D3 = 3 * D;          // 3072

// ---------------- vector types ----------------
typedef __attribute__((ext_vector_type(8)))  __bf16 v8bf;
typedef __attribute__((ext_vector_type(16))) __bf16 v16bf;
typedef __attribute__((ext_vector_type(8)))  float  v8f;
typedef __attribute__((ext_vector_type(4)))  unsigned int u32x4;
typedef __attribute__((ext_vector_type(8)))  int i32x8;
typedef __attribute__((ext_vector_type(4)))  int i32x4;

#define DEV static __device__ __forceinline__

// A-fragment (16x32 bf16, MxK) from row-major [rows, ld] bf16.
// lane: m = lane&15 (row), half = lane>>4.
// elements 0..7  = K (k0 + half*8 + e); 8..15 = K (k0 + 16 + half*8 + e)
DEV v16bf load_afrag(const __bf16* base, int ld, int row0, int k0, int lane) {
  const int m = lane & 15, hf = lane >> 4;
  const __bf16* p = base + (size_t)(row0 + m) * ld + k0 + hf * 8;
  v8bf lo = *(const v8bf*)(p);
  v8bf hi = *(const v8bf*)(p + 16);
  return __builtin_shufflevector(lo, hi, 0,1,2,3,4,5,6,7,8,9,10,11,12,13,14,15);
}

// B-fragment (32x16 bf16, KxN) from row-major weight [N, K] (op = W^T).
// lane: n = lane&15 (col), half = lane>>4; elements 0..7 = K (k0 + half*8 + e); 8..15 = K (k0 + 16 + half*8 + e)
DEV v16bf load_bfrag(const __bf16* base, int ld, int col0, int k0, int lane) {
  const int n = lane & 15, hf = lane >> 4;
  const __bf16* p = base + (size_t)(col0 + n) * ld + k0 + hf * 8;
  v8bf lo = *(const v8bf*)(p);
  v8bf hi = *(const v8bf*)(p + 16);
  return __builtin_shufflevector(lo, hi, 0,1,2,3,4,5,6,7,8,9,10,11,12,13,14,15);
}

DEV v8f wmma_bf16(v16bf a, v16bf b, v8f c) {
  return __builtin_amdgcn_wmma_f32_16x16x32_bf16(false, a, false, b, (short)0, c,
                                                 false, false);
}

DEV float red_max16(float v) {
  v = fmaxf(v, __shfl_xor(v, 1));
  v = fmaxf(v, __shfl_xor(v, 2));
  v = fmaxf(v, __shfl_xor(v, 4));
  v = fmaxf(v, __shfl_xor(v, 8));
  return v;
}
DEV float red_sum16(float v) {
  v += __shfl_xor(v, 1);
  v += __shfl_xor(v, 2);
  v += __shfl_xor(v, 4);
  v += __shfl_xor(v, 8);
  return v;
}

// ---------------- TDM: load a [nrows x 32] bf16 tile into LDS ----------------
// D# built per CDNA5 ISA.
// Padding: interval 16 DWORDs (one 64B row), amount 4 DWORDs (16B) ->
// LDS row stride = 40 bf16 (bank-conflict-free for ds_load_b128 B-fragments).
DEV void tdm_load_tile(const __bf16* gsrc, unsigned lds_off, int nrows, int ld_el) {
  unsigned long long ga = (unsigned long long)(uintptr_t)gsrc;
  u32x4 g0;
  g0[0] = 1u;                                          // count=1, user mode
  g0[1] = lds_off;                                     // lds_addr (bytes)
  g0[2] = (unsigned)(ga & 0xffffffffu);                // global_addr[31:0]
  g0[3] = (unsigned)((ga >> 32) & 0x01ffffffu) | (2u << 30);  // [56:32] | type=2
  const unsigned td0 = (unsigned)ld_el;                // tensor_dim0 (elements)
  const unsigned td1 = 0x00100000u;                    // tensor_dim1 (ample)
  const unsigned long long st0 = (unsigned long long)ld_el;  // dim0 stride (elems)
  i32x8 g1;
  g1[0] = (int)((1u << 16)          // data_size = 2 bytes
              | (1u << 20)          // pad_enable
              | (3u << 22)          // pad_interval: 16 DWORDs
              | (3u << 25));        // pad_amount: 4 DWORDs
  g1[1] = (int)((td0 & 0xffffu) << 16);
  g1[2] = (int)((td0 >> 16) | ((td1 & 0xffffu) << 16));
  g1[3] = (int)((td1 >> 16) | (32u << 16));            // tile_dim0 = 32
  g1[4] = (int)(unsigned)nrows;                        // tile_dim1; tile_dim2=0
  g1[5] = (int)(unsigned)(st0 & 0xffffffffu);
  g1[6] = (int)(unsigned)((st0 >> 32) & 0xffffu);      // stride0 hi; stride1 lo=0
  g1[7] = 0;
  i32x4 z4 = {0, 0, 0, 0};
#if defined(__clang_major__) && (__clang_major__ >= 23)
  i32x8 z8 = {0, 0, 0, 0, 0, 0, 0, 0};
  __builtin_amdgcn_tensor_load_to_lds(g0, g1, z4, z4, z8, 0);
#else
  __builtin_amdgcn_tensor_load_to_lds(g0, g1, z4, z4, 0);
#endif
}

// ---------------- embed: x = embed_w[ids] + pos_w ----------------
__global__ void embed_kernel(const int* __restrict__ ids,
                             const float* __restrict__ ew,
                             const float* __restrict__ pw,
                             float* __restrict__ x) {
  int idx = blockIdx.x * blockDim.x + threadIdx.x;    // over M * D/4
  if (idx >= M * (D / 4)) return;
  int bs = idx / (D / 4);
  int d4 = idx - bs * (D / 4);
  int s  = bs % S;
  int id = ids[bs];
  const float4 e = ((const float4*)ew)[(size_t)id * (D / 4) + d4];
  const float4 p = ((const float4*)pw)[(size_t)s  * (D / 4) + d4];
  float4 o; o.x = e.x + p.x; o.y = e.y + p.y; o.z = e.z + p.z; o.w = e.w + p.w;
  ((float4*)x)[idx] = o;
}

// ---------------- fp32 -> bf16 cast (weights) ----------------
struct alignas(8) bf16x4 { __bf16 a, b, c, d; };
__global__ void cast_kernel(const float* __restrict__ src, __bf16* __restrict__ dst,
                            int n4) {
  int idx = blockIdx.x * blockDim.x + threadIdx.x;
  if (idx >= n4) return;
  float4 v = ((const float4*)src)[idx];
  bf16x4 o; o.a = (__bf16)v.x; o.b = (__bf16)v.y; o.c = (__bf16)v.z; o.d = (__bf16)v.w;
  ((bf16x4*)dst)[idx] = o;
}

// ---------------- LayerNorm: wave per row, fp32 in -> bf16 out ---------------
__global__ __launch_bounds__(256) void ln_kernel(const float* __restrict__ x,
                                                 const float* __restrict__ sc,
                                                 const float* __restrict__ bi,
                                                 __bf16* __restrict__ out,
                                                 int rows, int dim) {
  const int lane = threadIdx.x & 31;
  const int wave = threadIdx.x >> 5;
  const int row  = blockIdx.x * 8 + wave;
  if (row >= rows) return;
  const float* xr = x + (size_t)row * dim;
  float sum = 0.f, sq = 0.f;
  for (int i = lane; i < dim; i += 32) {
    float v = xr[i];
    sum += v; sq += v * v;
  }
  #pragma unroll
  for (int m = 1; m < 32; m <<= 1) {
    sum += __shfl_xor(sum, m);
    sq  += __shfl_xor(sq, m);
  }
  const float mu   = sum / dim;
  const float var  = sq / dim - mu * mu;
  const float rstd = rsqrtf(var + 1e-5f);
  __bf16* orow = out + (size_t)row * dim;
  for (int i = lane; i < dim; i += 32)
    orow[i] = (__bf16)((xr[i] - mu) * rstd * sc[i] + bi[i]);
}

// ---------------- GEMM: C[M,N] = A[M,K] * W[N,K]^T + bias (+gelu)(+resid) ----
// block = 256 (8 waves); wave tile 16(M) x 64(N); block tile 128 x 64.
// Weight tile (64 x 32 bf16) staged in LDS by the TDM, double buffered;
// all 8 waves share it (8x cut in global weight traffic per block).
template <bool GELU, bool RESID, bool OUTBF16>
__global__ __launch_bounds__(256) void gemm_kernel(const __bf16* __restrict__ A,
                                                   const __bf16* __restrict__ W,
                                                   const float* __restrict__ bias,
                                                   const float* __restrict__ resid,
                                                   void* __restrict__ outp,
                                                   int Md, int Nd, int Kd) {
  __shared__ __align__(16) __bf16 wtile[2][64 * 40];   // 40 = 32 + TDM pad
  const int lane = threadIdx.x & 31;
  const int wave = threadIdx.x >> 5;
  const int row0 = blockIdx.y * 128 + wave * 16;
  const int col0 = blockIdx.x * 64;
  const unsigned lds0 = (unsigned)(uintptr_t)(&wtile[0][0]);
  const unsigned lds1 = (unsigned)(uintptr_t)(&wtile[1][0]);
  const __bf16* wblk = W + (size_t)col0 * Kd;
  const int nk = Kd >> 5;

  if (wave == 0) tdm_load_tile(wblk, lds0, 64, Kd);

  v8f acc[4] = {v8f{}, v8f{}, v8f{}, v8f{}};
  for (int i = 0; i < nk; ++i) {
    const int k0 = i << 5;
    if (wave == 0) __builtin_amdgcn_s_wait_tensorcnt(0);
    __syncthreads();                       // tile i visible; buf i^1 idle
    if (wave == 0 && (i + 1) < nk)
      tdm_load_tile(wblk + k0 + 32, (i & 1) ? lds0 : lds1, 64, Kd);
    if ((i + 1) < nk) {                    // prefetch next A fragment bytes
      const __bf16* pf = A + (size_t)(row0 + (lane & 15)) * Kd + (k0 + 32) +
                         (lane >> 4) * 8;
      __builtin_prefetch(pf, 0, 1);
    }
    const __bf16* wt = &wtile[i & 1][0];
    v16bf a = load_afrag(A, Kd, row0, k0, lane);
    #pragma unroll
    for (int t = 0; t < 4; ++t) {
      v16bf b = load_bfrag(wt, 40, t * 16, 0, lane);
      acc[t] = wmma_bf16(a, b, acc[t]);
    }
  }

  const int hf = lane >> 4, n = lane & 15;
  #pragma unroll
  for (int t = 0; t < 4; ++t) {
    const int col = col0 + t * 16 + n;
    const float bv = bias[col];
    #pragma unroll
    for (int r = 0; r < 8; ++r) {
      const int row = row0 + r + 8 * hf;
      float v = acc[t][r] + bv;
      if (GELU) v = 0.5f * v * (1.0f + erff(v * 0.70710678118654752f));
      if (RESID) v += resid[(size_t)row * Nd + col];
      if (OUTBF16) ((__bf16*)outp)[(size_t)row * Nd + col] = (__bf16)v;
      else         ((float*)outp)[(size_t)row * Nd + col] = v;
    }
  }
}

// ---------------- QKV repack: [M,3D] bf16 -> q,k [B,H,S,DH], vT [B,H,DH,S] ---
__global__ void repack_kernel(const __bf16* __restrict__ qkv,
                              __bf16* __restrict__ q, __bf16* __restrict__ k,
                              __bf16* __restrict__ vT) {
  int idx = blockIdx.x * blockDim.x + threadIdx.x;   // over M*D
  if (idx >= M * D) return;
  int m = idx / D, d = idx - m * D;
  int b = m / S, s = m - b * S;
  int h = d / DH, dh = d - h * DH;
  size_t bh = (size_t)(b * H + h);
  const __bf16* row = qkv + (size_t)m * D3;
  float qv = (float)row[d] * 0.125f;                 // 1/sqrt(DH)
  q [bh * S * DH + (size_t)s * DH + dh] = (__bf16)qv;
  k [bh * S * DH + (size_t)s * DH + dh] = row[D + d];
  vT[bh * DH * S + (size_t)dh * S + s] = row[2 * D + d];
}

// ---------------- flash attention: grid (S/64, H, B), block 128 -------------
__global__ __launch_bounds__(128) void attn_kernel(const __bf16* __restrict__ q,
                                                   const __bf16* __restrict__ k,
                                                   const __bf16* __restrict__ vT,
                                                   __bf16* __restrict__ obf) {
  __shared__ __align__(16) __bf16 plds[4][16][40];
  const int lane = threadIdx.x & 31;
  const int wave = threadIdx.x >> 5;
  const int h = blockIdx.y, b = blockIdx.z;
  const int q0 = blockIdx.x * 64 + wave * 16;
  const __bf16* qb = q  + (size_t)(b * H + h) * S * DH;
  const __bf16* kb = k  + (size_t)(b * H + h) * S * DH;
  const __bf16* vb = vT + (size_t)(b * H + h) * DH * S;

  const v16bf aQ0 = load_afrag(qb, DH, q0, 0, lane);
  const v16bf aQ1 = load_afrag(qb, DH, q0, 32, lane);
  v8f o0 = {}, o1 = {}, o2 = {}, o3 = {};
  float mrow[8], lrow[8];
  #pragma unroll
  for (int r = 0; r < 8; ++r) { mrow[r] = -1e30f; lrow[r] = 0.f; }
  const int hf = lane >> 4, n = lane & 15;

  for (int j0 = 0; j0 < S; j0 += 32) {
    v8f c0 = {}, c1 = {};
    c0 = wmma_bf16(aQ0, load_bfrag(kb, DH, j0,       0,  lane), c0);
    c0 = wmma_bf16(aQ1, load_bfrag(kb, DH, j0,       32, lane), c0);
    c1 = wmma_bf16(aQ0, load_bfrag(kb, DH, j0 + 16,  0,  lane), c1);
    c1 = wmma_bf16(aQ1, load_bfrag(kb, DH, j0 + 16,  32, lane), c1);
    #pragma unroll
    for (int r = 0; r < 8; ++r) {
      float mx   = red_max16(fmaxf(c0[r], c1[r]));
      float mnew = fmaxf(mrow[r], mx);
      float scl  = __expf(mrow[r] - mnew);
      o0[r] *= scl; o1[r] *= scl; o2[r] *= scl; o3[r] *= scl;
      float p0 = __expf(c0[r] - mnew);
      float p1 = __expf(c1[r] - mnew);
      lrow[r] = lrow[r] * scl + red_sum16(p0 + p1);
      mrow[r] = mnew;
      const int row = r + 8 * hf;
      plds[wave][row][n]      = (__bf16)p0;
      plds[wave][row][16 + n] = (__bf16)p1;
    }
    v16bf aP = load_afrag(&plds[wave][0][0], 40, 0, 0, lane);
    o0 = wmma_bf16(aP, load_bfrag(vb, S, 0,  j0, lane), o0);
    o1 = wmma_bf16(aP, load_bfrag(vb, S, 16, j0, lane), o1);
    o2 = wmma_bf16(aP, load_bfrag(vb, S, 32, j0, lane), o2);
    o3 = wmma_bf16(aP, load_bfrag(vb, S, 48, j0, lane), o3);
  }
  #pragma unroll
  for (int r = 0; r < 8; ++r) {
    const float inv = 1.0f / lrow[r];
    __bf16* orow = obf + ((size_t)(b * S + q0 + r + 8 * hf)) * D + h * DH;
    orow[n]      = (__bf16)(o0[r] * inv);
    orow[16 + n] = (__bf16)(o1[r] * inv);
    orow[32 + n] = (__bf16)(o2[r] * inv);
    orow[48 + n] = (__bf16)(o3[r] * inv);
  }
}

// ---------------- host orchestration ----------------
extern "C" void kernel_launch(void* const* d_in, const int* in_sizes, int n_in,
                              void* d_out, int out_size, void* d_ws, size_t ws_size,
                              hipStream_t stream) {
  const int*   ids   = (const int*)  d_in[0];
  const float* ew    = (const float*)d_in[1];
  const float* pw    = (const float*)d_in[2];
  const float* ln1s  = (const float*)d_in[3];
  const float* ln1b  = (const float*)d_in[4];
  const float* wqkv  = (const float*)d_in[5];
  const float* bqkv  = (const float*)d_in[6];
  const float* wout  = (const float*)d_in[7];
  const float* bout  = (const float*)d_in[8];
  const float* ln2s  = (const float*)d_in[9];
  const float* ln2b  = (const float*)d_in[10];
  const float* wfc1  = (const float*)d_in[11];
  const float* bfc1  = (const float*)d_in[12];
  const float* wfc2  = (const float*)d_in[13];
  const float* bfc2  = (const float*)d_in[14];
  const float* lnfs  = (const float*)d_in[15];
  const float* lnfb  = (const float*)d_in[16];
  const float* wlm   = (const float*)d_in[17];
  const float* blm   = (const float*)d_in[18];

  char* ws = (char*)d_ws; size_t off = 0;
  auto carve = [&](size_t bytes) -> void* {
    void* p = ws + off; off += (bytes + 255) & ~(size_t)255; return p;
  };
  float*  x     = (float*) carve((size_t)M * D * 4);
  __bf16* hbf   = (__bf16*)carve((size_t)M * D * 2);
  __bf16* qkvbf = (__bf16*)carve((size_t)M * D3 * 2);
  __bf16* qb    = (__bf16*)carve((size_t)M * D * 2);
  __bf16* kbuf  = (__bf16*)carve((size_t)M * D * 2);
  __bf16* vTb   = (__bf16*)carve((size_t)M * D * 2);
  __bf16* obf   = (__bf16*)carve((size_t)M * D * 2);
  __bf16* h1bf  = (__bf16*)carve((size_t)M * F * 2);
  __bf16* wbuf  = (__bf16*)carve((size_t)V * D * 2);   // reused per weight

  auto cast_to = [&](const float* src, __bf16* dst, size_t n) {
    int n4 = (int)(n / 4);
    cast_kernel<<<(n4 + 255) / 256, 256, 0, stream>>>(src, dst, n4);
  };

  embed_kernel<<<(M * (D / 4) + 255) / 256, 256, 0, stream>>>(ids, ew, pw, x);

  const dim3 blk(256);
  for (int l = 0; l < L; ++l) {
    // --- attention block ---
    ln_kernel<<<M / 8, blk, 0, stream>>>(x, ln1s + (size_t)l * D, ln1b + (size_t)l * D,
                                         hbf, M, D);
    cast_to(wqkv + (size_t)l * D3 * D, wbuf, (size_t)D3 * D);
    gemm_kernel<false, false, true><<<dim3(D3 / 64, M / 128), blk, 0, stream>>>(
        hbf, wbuf, bqkv + (size_t)l * D3, nullptr, qkvbf, M, D3, D);
    repack_kernel<<<(M * D + 255) / 256, 256, 0, stream>>>(qkvbf, qb, kbuf, vTb);
    attn_kernel<<<dim3(S / 64, H, B), dim3(128), 0, stream>>>(qb, kbuf, vTb, obf);
    cast_to(wout + (size_t)l * D * D, wbuf, (size_t)D * D);
    gemm_kernel<false, true, false><<<dim3(D / 64, M / 128), blk, 0, stream>>>(
        obf, wbuf, bout + (size_t)l * D, x, x, M, D, D);
    // --- MLP block ---
    ln_kernel<<<M / 8, blk, 0, stream>>>(x, ln2s + (size_t)l * D, ln2b + (size_t)l * D,
                                         hbf, M, D);
    cast_to(wfc1 + (size_t)l * F * D, wbuf, (size_t)F * D);
    gemm_kernel<true, false, true><<<dim3(F / 64, M / 128), blk, 0, stream>>>(
        hbf, wbuf, bfc1 + (size_t)l * F, nullptr, h1bf, M, F, D);
    cast_to(wfc2 + (size_t)l * D * F, wbuf, (size_t)D * F);
    gemm_kernel<false, true, false><<<dim3(D / 64, M / 128), blk, 0, stream>>>(
        h1bf, wbuf, bfc2 + (size_t)l * D, x, x, M, D, F);
  }
  // --- final LN + LM head ---
  ln_kernel<<<M / 8, blk, 0, stream>>>(x, lnfs, lnfb, hbf, M, D);
  cast_to(wlm, wbuf, (size_t)V * D);
  gemm_kernel<false, false, false><<<dim3(V / 64, M / 128), blk, 0, stream>>>(
      hbf, wbuf, blm, nullptr, (float*)d_out, M, V, D);
}
